// TemporalBlock_39307540693322
// MI455X (gfx1250) — compile-verified, matrix operand order checked by reading, not run on hardware
//
#include <hip/hip_runtime.h>
#include <hip/hip_bf16.h>

typedef __attribute__((ext_vector_type(16))) __bf16 v16bf;
typedef __attribute__((ext_vector_type(8)))  float  v8f;

#define B_  32
#define C_  256
#define T_  1024
#define KS_ 64
#define PADT_ (T_ + 2)   // 2 zero rows in front for causal conv left-pad

static __device__ __forceinline__ v8f wmma_bf16(v16bf a, v16bf b, v8f c) {
  return __builtin_amdgcn_wmma_f32_16x16x32_bf16(false, a, false, b, (short)0, c, false, false);
}

// Fragment loader for sources where K is contiguous per row: element (r, kk) at
// base[r*stride + kk], r = lane&15 (M for A-frags, N for B-frags from a transposed
// view).  K packing per CDNA5 ISA 16-bit layout:
//   lanes 0-15: K {0..7} in v0..v3, {16..23} in v4..v7 ; lanes 16-31: +8.
// Per lane this is two contiguous 16-byte runs -> vectorizes to 2x global_load_b128.
static __device__ __forceinline__ v16bf load_frag_kcontig(const __bf16* base, int stride, int lane) {
  int r  = lane & 15;
  int kh = (lane >> 4) & 1;
  const __bf16* p = base + r * stride;
  v16bf f;
#pragma unroll
  for (int j = 0; j < 8; ++j) {
    int kk = ((j >> 2) << 4) + (kh << 3) + ((j & 3) << 1);
    f[2 * j]     = p[kk];
    f[2 * j + 1] = p[kk + 1];
  }
  return f;
}

// ---------------- prep kernels ----------------

// dst[n][c] = src[c][n]  (bf16 transposed weights so B-frags are K-contiguous)
__global__ void cvt_w_transpose(const float* __restrict__ src, __bf16* __restrict__ dst, int N) {
  int i = blockIdx.x * blockDim.x + threadIdx.x;
  if (i >= C_ * N) return;
  int n = i % N, c = i / N;
  dst[(size_t)n * C_ + c] = (__bf16)src[i];
}

__global__ void x_transpose_bf16(const float* __restrict__ x, __bf16* __restrict__ xtb) {
  int i = blockIdx.x * blockDim.x + threadIdx.x;   // over B*C*T in x layout
  if (i >= B_ * C_ * T_) return;
  int t = i % T_;
  int c = (i / T_) % C_;
  int b = i / (C_ * T_);
  xtb[((size_t)b * T_ + t) * C_ + c] = (__bf16)x[i];
}

// zero the 2 causal pad rows of both conv activation buffers
__global__ void zero_pad_rows(__bf16* __restrict__ h1, __bf16* __restrict__ h2) {
  int i = blockIdx.x * blockDim.x + threadIdx.x;   // over B*2*C
  if (i >= B_ * 2 * C_) return;
  int c = i % C_;
  int row = (i / C_) % 2;
  int b = i / (2 * C_);
  size_t idx = ((size_t)b * PADT_ + row) * C_ + c;
  h1[idx] = (__bf16)0.0f;
  h2[idx] = (__bf16)0.0f;
}

// weight_norm(dim=0): W = g * v / ||v||  (norm over in,k);  store as Wt[tap][o][i] bf16
__global__ __launch_bounds__(256) void weightnorm_bf16(const float* __restrict__ v,
                                                       const float* __restrict__ g,
                                                       __bf16* __restrict__ Wt) {
  __shared__ float red[256];
  int o = blockIdx.x, tid = threadIdx.x;
  float s = 0.f;
  for (int e = tid; e < C_ * 3; e += 256) { float w = v[o * C_ * 3 + e]; s += w * w; }
  red[tid] = s; __syncthreads();
  for (int st = 128; st > 0; st >>= 1) { if (tid < st) red[tid] += red[tid + st]; __syncthreads(); }
  float scale = g[o] / sqrtf(red[0]);
  for (int e = tid; e < C_ * 3; e += 256) {
    int i = e / 3, tap = e % 3;
    Wt[((size_t)tap * C_ + o) * C_ + i] = (__bf16)(scale * v[o * C_ * 3 + e]);
  }
}

// ---------------- QKV projection ----------------
// out[b,t,n] = xt[b,t,:] @ w[:,n] + bias[n];  wT is [N][C] so both operands K-contiguous.
// transposed_out=0: out[b][t][N] (q,k);  =1: out[b][N][t] (v, channel-major, contiguous stores)

__global__ __launch_bounds__(32) void linear_wmma(const __bf16* __restrict__ xtb,
                                                  const __bf16* __restrict__ wT,
                                                  const float* __restrict__ bias,
                                                  __bf16* __restrict__ out,
                                                  int N, int transposed_out) {
  int ns = blockIdx.x, tt = blockIdx.y, b = blockIdx.z;
  int lane = threadIdx.x;
  v8f zero = {};
  v8f acc[4] = {zero, zero, zero, zero};
  const __bf16* xb = xtb + (size_t)b * T_ * C_ + tt * 16 * C_;
#pragma unroll
  for (int cb = 0; cb < 8; ++cb) {
    int c0 = cb * 32;
    v16bf a = load_frag_kcontig(xb + c0, C_, lane);
#pragma unroll
    for (int ci = 0; ci < 4; ++ci) {
      v16bf bf = load_frag_kcontig(wT + (size_t)(ns * 64 + ci * 16) * C_ + c0, C_, lane);
      acc[ci] = wmma_bf16(a, bf, acc[ci]);
    }
  }
  int n = lane & 15, mh = (lane >> 4) * 8;
#pragma unroll
  for (int ci = 0; ci < 4; ++ci) {
    int col = ns * 64 + ci * 16 + n;
    float bb = bias[col];
#pragma unroll
    for (int r = 0; r < 8; ++r) {
      int row = tt * 16 + mh + r;
      float val = acc[ci][r] + bb;
      if (transposed_out)
        out[((size_t)b * N + col) * T_ + row] = (__bf16)val;   // per-lane contiguous in row
      else
        out[((size_t)b * T_ + row) * N + col] = (__bf16)val;
    }
  }
}

// ---------------- attention pass 1: per-(b,key) column max & sumexp over queries ----------------

__global__ __launch_bounds__(32) void attn_stats(const __bf16* __restrict__ qb,
                                                 const __bf16* __restrict__ kb,
                                                 float* __restrict__ colmax,
                                                 float* __restrict__ colsum) {
  int kt = blockIdx.x, b = blockIdx.y;
  int lane = threadIdx.x;
  const __bf16* qB = qb + (size_t)b * T_ * KS_;
  const __bf16* kB = kb + (size_t)b * T_ * KS_;
  int k0 = kt * 16;
  v16bf bk0 = load_frag_kcontig(kB + k0 * KS_ + 0,  KS_, lane);   // K^T tile, cols=keys
  v16bf bk1 = load_frag_kcontig(kB + k0 * KS_ + 32, KS_, lane);
  int n = lane & 15, mh = (lane >> 4) * 8;
  float m_run = -3.0e38f, s_run = 0.f;
  for (int qt = kt; qt < T_ / 16; ++qt) {                         // causal: skip qt<kt
    v16bf a0 = load_frag_kcontig(qB + qt * 16 * KS_ + 0,  KS_, lane);
    v16bf a1 = load_frag_kcontig(qB + qt * 16 * KS_ + 32, KS_, lane);
    v8f acc = {};
    acc = wmma_bf16(a0, bk0, acc);
    acc = wmma_bf16(a1, bk1, acc);
    float z[8];
    float tmax = -3.0e38f;
#pragma unroll
    for (int r = 0; r < 8; ++r) {
      int qq = qt * 16 + mh + r;
      float zz = (qq < k0 + n) ? -3.0e38f : acc[r] * 0.125f;      // /sqrt(64), key>query masked
      z[r] = zz;
      tmax = fmaxf(tmax, zz);
    }
    tmax = fmaxf(tmax, __shfl_xor(tmax, 16, 32));                 // combine row halves of column
    float ts = 0.f;
#pragma unroll
    for (int r = 0; r < 8; ++r)
      if (z[r] > -1.0e37f) ts += __expf(z[r] - tmax);
    ts += __shfl_xor(ts, 16, 32);
    float mnew = fmaxf(m_run, tmax);
    s_run = s_run * __expf(m_run - mnew) + ts * __expf(tmax - mnew);
    m_run = mnew;
  }
  if (lane < 16) {
    colmax[(size_t)b * T_ + k0 + lane] = m_run;
    colsum[(size_t)b * T_ + k0 + lane] = s_run;
  }
}

// ---------------- attention pass 2: out_attn = P @ V (P transposed via LDS), rowsum(P) ----------------
// vT is channel-major [B][C][T] so V B-frags are K-contiguous over the key index.

__global__ __launch_bounds__(32) void attn_out(const __bf16* __restrict__ qb,
                                               const __bf16* __restrict__ kb,
                                               const __bf16* __restrict__ vT,
                                               const float* __restrict__ colmax,
                                               const float* __restrict__ colsum,
                                               __bf16* __restrict__ h1,
                                               float* __restrict__ rowsum) {
  __shared__ __bf16 ldsP[16][32];           // 16 query rows x 32 key cols (one K-group)
  int cs = blockIdx.x, qt = blockIdx.y, b = blockIdx.z;
  int lane = threadIdx.x;
  int n = lane & 15, mh = (lane >> 4) * 8;
  const __bf16* qB = qb + (size_t)b * T_ * KS_;
  const __bf16* kB = kb + (size_t)b * T_ * KS_;
  v16bf a0 = load_frag_kcontig(qB + qt * 16 * KS_ + 0,  KS_, lane);
  v16bf a1 = load_frag_kcontig(qB + qt * 16 * KS_ + 32, KS_, lane);
  v8f zero = {};
  v8f acc[4] = {zero, zero, zero, zero};
  float rs = 0.f;
  int npair = (qt >> 1) + 1;                // key tiles 0..qt, processed 2 at a time
  for (int pp = 0; pp < npair; ++pp) {
#pragma unroll
    for (int half = 0; half < 2; ++half) {
      int kt = 2 * pp + half;
      if (kt <= qt) {                       // wave-uniform branch
        int k0 = kt * 16;
        v16bf bk0 = load_frag_kcontig(kB + k0 * KS_ + 0,  KS_, lane);
        v16bf bk1 = load_frag_kcontig(kB + k0 * KS_ + 32, KS_, lane);
        v8f s = {};
        s = wmma_bf16(a0, bk0, s);
        s = wmma_bf16(a1, bk1, s);
        float mk = colmax[(size_t)b * T_ + k0 + n];
        float rz = 1.0f / colsum[(size_t)b * T_ + k0 + n];
#pragma unroll
        for (int r = 0; r < 8; ++r) {
          int qq = qt * 16 + mh + r;
          float p = (qq < k0 + n) ? 0.f : __expf(s[r] * 0.125f - mk) * rz;
          ldsP[mh + r][half * 16 + n] = (__bf16)p;
        }
      } else {
#pragma unroll
        for (int r = 0; r < 8; ++r) ldsP[mh + r][half * 16 + n] = (__bf16)0.0f;
      }
    }
    __syncthreads();
    if (cs == 0 && lane < 16) {             // accumulate sum_k attn for weight_x
#pragma unroll
      for (int cc = 0; cc < 32; ++cc) rs += (float)ldsP[lane][cc];
    }
    v16bf pa;                               // P tile as A-fragment; rows are 64B, 16B-aligned runs
    {
      int r = lane & 15, kh = (lane >> 4) & 1;
#pragma unroll
      for (int j = 0; j < 8; ++j) {
        int kk = ((j >> 2) << 4) + (kh << 3) + ((j & 3) << 1);
        pa[2 * j]     = ldsP[r][kk];
        pa[2 * j + 1] = ldsP[r][kk + 1];
      }
    }
#pragma unroll
    for (int ci = 0; ci < 4; ++ci) {
      v16bf bvf = load_frag_kcontig(vT + ((size_t)b * C_ + cs * 64 + ci * 16) * T_ + pp * 32, T_, lane);
      acc[ci] = wmma_bf16(pa, bvf, acc[ci]);
    }
    __syncthreads();
  }
#pragma unroll
  for (int ci = 0; ci < 4; ++ci) {
#pragma unroll
    for (int r = 0; r < 8; ++r) {
      int c = cs * 64 + ci * 16 + n;
      int t = qt * 16 + mh + r;
      h1[((size_t)b * PADT_ + t + 2) * C_ + c] = (__bf16)acc[ci][r];  // [B][T+2][C] for conv
    }
  }
  if (cs == 0 && lane < 16)
    rowsum[(size_t)b * T_ + qt * 16 + lane] = rs;
}

// ---------------- weight_x = softmax_t(rowsum) ----------------

__global__ __launch_bounds__(256) void wx_softmax(const float* __restrict__ rowsum,
                                                  float* __restrict__ wx) {
  __shared__ float red[256];
  int b = blockIdx.x, tid = threadIdx.x;
  const float* rb = rowsum + (size_t)b * T_;
  float m = -3.0e38f;
  for (int t = tid; t < T_; t += 256) m = fmaxf(m, rb[t]);
  red[tid] = m; __syncthreads();
  for (int st = 128; st > 0; st >>= 1) { if (tid < st) red[tid] = fmaxf(red[tid], red[tid + st]); __syncthreads(); }
  float mm = red[0]; __syncthreads();
  float s = 0.f;
  for (int t = tid; t < T_; t += 256) s += __expf(rb[t] - mm);
  red[tid] = s; __syncthreads();
  for (int st = 128; st > 0; st >>= 1) { if (tid < st) red[tid] += red[tid + st]; __syncthreads(); }
  float inv = 1.0f / red[0];
  for (int t = tid; t < T_; t += 256) wx[(size_t)b * T_ + t] = __expf(rb[t] - mm) * inv;
}

// ---------------- causal conv (3-tap GEMM over channels) ----------------
// h is [B][T+2][C] (2 zeroed pad rows).  The 66-row input strip is staged into LDS
// with gfx1250 async global->LDS copies, then B-frags read K-contiguous from LDS.

__global__ __launch_bounds__(32) void conv_wmma(const __bf16* __restrict__ h,
                                                const __bf16* __restrict__ Wt,
                                                const float* __restrict__ bias,
                                                __bf16* __restrict__ out_bf,
                                                float* __restrict__ out_f,
                                                const float* __restrict__ x,
                                                const float* __restrict__ wx,
                                                int fuse) {
  __shared__ __bf16 ldsH[66 * C_];          // 66 rows x 256 ch x 2B = 33 KB
  int ts = blockIdx.x, ot = blockIdx.y, b = blockIdx.z;
  int lane = threadIdx.x;
  int n = lane & 15, mh = (lane >> 4) * 8;

  // ---- async stage: padded rows [ts*64, ts*64+66) of batch b into LDS ----
  {
    const char* src = (const char*)(h + ((size_t)b * PADT_ + ts * 64) * C_);
    unsigned ldsbase = (unsigned)(size_t)(void*)ldsH;   // LDS aperture: low 32 bits = LDS offset
    for (int it = 0; it < 66; ++it) {                   // 66 * 32 lanes * 16B = 33792 B
      unsigned off = (unsigned)(it * 32 + lane) * 16u;
      unsigned long long ga = (unsigned long long)src + off;
      unsigned la = ldsbase + off;
      asm volatile("global_load_async_to_lds_b128 %0, %1, off"
                   :: "v"(la), "v"(ga) : "memory");
    }
    asm volatile("s_wait_asynccnt 0" ::: "memory");
  }
  __syncthreads();

  v8f zero = {};
  v8f acc[4] = {zero, zero, zero, zero};
  for (int tap = 0; tap < 3; ++tap) {
    for (int ib = 0; ib < 8; ++ib) {
      int i0 = ib * 32;
      v16bf a = load_frag_kcontig(Wt + ((size_t)tap * C_ + ot * 16) * C_ + i0, C_, lane);
#pragma unroll
      for (int ci = 0; ci < 4; ++ci) {
        // local padded row within strip = ci*16 + tap + n  (0..65, never OOB thanks to pad)
        int rowbase = (ci * 16 + tap + n) * C_ + i0;
        v16bf bf;
        int kh = (lane >> 4) & 1;
#pragma unroll
        for (int j = 0; j < 8; ++j) {
          int kk = ((j >> 2) << 4) + (kh << 3) + ((j & 3) << 1);
          bf[2 * j]     = ldsH[rowbase + kk];
          bf[2 * j + 1] = ldsH[rowbase + kk + 1];
        }
        acc[ci] = wmma_bf16(a, bf, acc[ci]);
      }
    }
  }
#pragma unroll
  for (int ci = 0; ci < 4; ++ci) {
#pragma unroll
    for (int r = 0; r < 8; ++r) {
      int o = ot * 16 + mh + r;
      int t = ts * 64 + ci * 16 + n;
      float y = fmaxf(acc[ci][r] + bias[o], 0.f);
      if (!fuse) {
        out_bf[((size_t)b * PADT_ + t + 2) * C_ + o] = (__bf16)y;  // per-lane contiguous in o
      } else {
        size_t idx = ((size_t)b * C_ + o) * T_ + t;
        float xx = x[idx];
        out_f[idx] = fmaxf(y + xx + wx[(size_t)b * T_ + t] * xx, 0.f);
      }
    }
  }
}

// ---------------- launch ----------------

extern "C" void kernel_launch(void* const* d_in, const int* in_sizes, int n_in,
                              void* d_out, int out_size, void* d_ws, size_t ws_size,
                              hipStream_t stream) {
  (void)in_sizes; (void)n_in; (void)out_size; (void)ws_size;
  const float* x  = (const float*)d_in[0];
  const float* wq = (const float*)d_in[1];
  const float* bq = (const float*)d_in[2];
  const float* wk = (const float*)d_in[3];
  const float* bk = (const float*)d_in[4];
  const float* wv = (const float*)d_in[5];
  const float* bv = (const float*)d_in[6];
  const float* v1 = (const float*)d_in[7];
  const float* g1 = (const float*)d_in[8];
  const float* b1 = (const float*)d_in[9];
  const float* v2 = (const float*)d_in[10];
  const float* g2 = (const float*)d_in[11];
  const float* b2 = (const float*)d_in[12];
  float* out = (float*)d_out;

  char* ws = (char*)d_ws;
  size_t off = 0;
  auto take = [&](size_t bytes) -> char* {
    char* p = ws + off;
    off += bytes;
    off = (off + 255) & ~(size_t)255;
    return p;
  };
  __bf16* qb   = (__bf16*)take((size_t)B_ * T_ * KS_ * 2);
  __bf16* kbuf = (__bf16*)take((size_t)B_ * T_ * KS_ * 2);
  __bf16* vT   = (__bf16*)take((size_t)B_ * C_ * T_ * 2);       // channel-major V
  __bf16* xtb  = (__bf16*)take((size_t)B_ * T_ * C_ * 2);
  __bf16* h1   = (__bf16*)take((size_t)B_ * PADT_ * C_ * 2);    // padded conv activations
  __bf16* h2   = (__bf16*)take((size_t)B_ * PADT_ * C_ * 2);
  __bf16* wqT  = (__bf16*)take((size_t)KS_ * C_ * 2);
  __bf16* wkT  = (__bf16*)take((size_t)KS_ * C_ * 2);
  __bf16* wvT  = (__bf16*)take((size_t)C_ * C_ * 2);
  __bf16* Wt1  = (__bf16*)take((size_t)3 * C_ * C_ * 2);
  __bf16* Wt2  = (__bf16*)take((size_t)3 * C_ * C_ * 2);
  float* colmax = (float*)take((size_t)B_ * T_ * 4);
  float* colsum = (float*)take((size_t)B_ * T_ * 4);
  float* rowsum = (float*)take((size_t)B_ * T_ * 4);
  float* wxv    = (float*)take((size_t)B_ * T_ * 4);

  cvt_w_transpose<<<(C_ * KS_ + 255) / 256, 256, 0, stream>>>(wq, wqT, KS_);
  cvt_w_transpose<<<(C_ * KS_ + 255) / 256, 256, 0, stream>>>(wk, wkT, KS_);
  cvt_w_transpose<<<(C_ * C_ + 255) / 256, 256, 0, stream>>>(wv, wvT, C_);
  weightnorm_bf16<<<C_, 256, 0, stream>>>(v1, g1, Wt1);
  weightnorm_bf16<<<C_, 256, 0, stream>>>(v2, g2, Wt2);
  x_transpose_bf16<<<(B_ * C_ * T_ + 255) / 256, 256, 0, stream>>>(x, xtb);
  zero_pad_rows<<<(B_ * 2 * C_ + 255) / 256, 256, 0, stream>>>(h1, h2);

  linear_wmma<<<dim3(1, T_ / 16, B_), 32, 0, stream>>>(xtb, wqT, bq, qb, KS_, 0);
  linear_wmma<<<dim3(1, T_ / 16, B_), 32, 0, stream>>>(xtb, wkT, bk, kbuf, KS_, 0);
  linear_wmma<<<dim3(C_ / 64, T_ / 16, B_), 32, 0, stream>>>(xtb, wvT, bv, vT, C_, 1);

  attn_stats<<<dim3(T_ / 16, B_), 32, 0, stream>>>(qb, kbuf, colmax, colsum);
  attn_out<<<dim3(C_ / 64, T_ / 16, B_), 32, 0, stream>>>(qb, kbuf, vT, colmax, colsum, h1, rowsum);
  wx_softmax<<<B_, 256, 0, stream>>>(rowsum, wxv);

  conv_wmma<<<dim3(T_ / 64, C_ / 16, B_), 32, 0, stream>>>(h1, Wt1, b1, h2, nullptr, nullptr, nullptr, 0);
  conv_wmma<<<dim3(T_ / 64, C_ / 16, B_), 32, 0, stream>>>(h2, Wt2, b2, nullptr, out, x, wxv, 1);
}
